// TransformerBlock_73555609911370
// MI455X (gfx1250) — matrix operand order repaired and checked by reading, not compiled, not run on hardware
//
#include <hip/hip_runtime.h>
#include <hip/hip_bf16.h>
#include <math.h>

#define EMB   1024
#define NHEAD 16
#define HEADD 64
#define FFD   4096
#define BBATCH 4
#define TSEQ  2048
#define BT    (BBATCH*TSEQ)

typedef __attribute__((ext_vector_type(16))) __bf16 v16bf;
typedef __attribute__((ext_vector_type(8)))  __bf16 v8bf;
typedef __attribute__((ext_vector_type(8)))  float  v8f;
typedef int v4i __attribute__((vector_size(16)));

// ---------------- CDNA5 async global->LDS copy (guarded, with sync fallback) ----

#if __has_builtin(__builtin_amdgcn_global_load_async_to_lds_b128) && \
    __has_builtin(__builtin_amdgcn_s_wait_asynccnt)
#define HAVE_ASYNC 1
#else
#define HAVE_ASYNC 0
#endif

__device__ __forceinline__ void copy16(__bf16* dst_lds, const __bf16* src_glob) {
#if HAVE_ASYNC
    __builtin_amdgcn_global_load_async_to_lds_b128(
        (__attribute__((address_space(1))) v4i*)src_glob,
        (__attribute__((address_space(3))) v4i*)dst_lds,
        0, 0);
#else
    *(uint4*)dst_lds = *(const uint4*)src_glob;
#endif
}

__device__ __forceinline__ void copy_wait() {
#if HAVE_ASYNC
    __builtin_amdgcn_s_wait_asynccnt(0);
#endif
}

// ---------------- helpers ----------------

__device__ __forceinline__ __bf16 f2bf(float f) {
    unsigned u = __builtin_bit_cast(unsigned, f);
    u += 0x7fffu + ((u >> 16) & 1u);              // round-to-nearest-even
    unsigned short s = (unsigned short)(u >> 16);
    return __builtin_bit_cast(__bf16, s);
}

// Build a 16-element bf16 fragment from two 16B runs (each 16B-aligned).
__device__ __forceinline__ v16bf ld_frag2(const __bf16* lo, const __bf16* hi) {
    v8bf a = *(const v8bf*)lo;
    v8bf b = *(const v8bf*)hi;
    v16bf r;
#pragma unroll
    for (int i = 0; i < 8; ++i) { r[i] = a[i]; r[i + 8] = b[i]; }
    return r;
}
__device__ __forceinline__ v16bf ld_frag(const __bf16* p) { return ld_frag2(p, p + 8); }

__device__ __forceinline__ v8f wmma_bf16(v16bf a, v16bf b, v8f c) {
    return __builtin_amdgcn_wmma_f32_16x16x32_bf16(false, a, false, b, (short)0, c, false, false);
}

// ---------------- weight convert + transpose: wT[n][k] = bf16(w[k][n]) ----------------

__global__ __launch_bounds__(256) void wconv_t(const float* __restrict__ w,
                                               __bf16* __restrict__ wT, int K, int N) {
    size_t i = (size_t)blockIdx.x * 256 + threadIdx.x;
    if (i >= (size_t)K * N) return;
    int k = (int)(i / N);
    int n = (int)(i % N);
    wT[(size_t)n * K + k] = f2bf(w[i]);
}

// ---------------- LayerNorm (row=1024) -> bf16 ----------------

__global__ __launch_bounds__(256) void ln_kernel(const float* __restrict__ x,
                                                 const float* __restrict__ w,
                                                 const float* __restrict__ b,
                                                 __bf16* __restrict__ out) {
    __shared__ float rs[8], rs2[8];
    __shared__ float mu_s, rstd_s;
    int r = blockIdx.x;
    int t = threadIdx.x;
    const float4 xv = *(const float4*)(x + (size_t)r * EMB + t * 4);
    float s  = xv.x + xv.y + xv.z + xv.w;
    float s2 = xv.x * xv.x + xv.y * xv.y + xv.z * xv.z + xv.w * xv.w;
#pragma unroll
    for (int off = 16; off > 0; off >>= 1) {
        s  += __shfl_xor(s,  off, 32);
        s2 += __shfl_xor(s2, off, 32);
    }
    int wid = t >> 5, lane = t & 31;
    if (lane == 0) { rs[wid] = s; rs2[wid] = s2; }
    __syncthreads();
    if (t == 0) {
        float S = 0.f, S2 = 0.f;
#pragma unroll
        for (int i = 0; i < 8; ++i) { S += rs[i]; S2 += rs2[i]; }
        float mu  = S * (1.0f / EMB);
        float var = S2 * (1.0f / EMB) - mu * mu;
        mu_s = mu;
        rstd_s = rsqrtf(var + 1e-6f);
    }
    __syncthreads();
    float mu = mu_s, rstd = rstd_s;
    const float xs[4] = {xv.x, xv.y, xv.z, xv.w};
#pragma unroll
    for (int i = 0; i < 4; ++i) {
        int c = t * 4 + i;
        out[(size_t)r * EMB + c] = f2bf((xs[i] - mu) * rstd * w[c] + b[c]);
    }
}

// ---------------- bf16 GEMM: C = A[M,K] * Bt[N,K]^T, 256x128 block tile ----------------
// Double-buffered LDS pipeline with async global->LDS staging.
// EPI: 0 = QKV scatter (q scaled by 0.125), 1 = proj+bias+residual->f32,
//      2 = fc1+bias+GELU->bf16, 3 = fc2+bias+residual->f32

template <int EPI>
__global__ __launch_bounds__(256) void gemm_bf16(const __bf16* __restrict__ A,
                                                 const __bf16* __restrict__ Bt,
                                                 const float* __restrict__ bias,
                                                 const float* __restrict__ resid,
                                                 float* __restrict__ outf,
                                                 __bf16* __restrict__ outb,
                                                 __bf16* __restrict__ qv,
                                                 __bf16* __restrict__ kv,
                                                 __bf16* __restrict__ vv,
                                                 int M, int N, int K) {
    __shared__ __bf16 As[2][256][40];   // 2 x 256x32 tiles (+8 pad, row stride 80B)
    __shared__ __bf16 Bs[2][128][40];   // 2 x 128x32 tiles

    const int t    = threadIdx.x;
    const int w    = t >> 5;
    const int lane = t & 31;
    const int g    = lane >> 4;
    const int cl   = lane & 15;

    const int m0 = blockIdx.y * 256;
    const int n0 = blockIdx.x * 128;

    const int rowB  = t >> 1;
    const int halfB = (t & 1) * 16;

    const __bf16* gA = A  + (size_t)(m0 + t) * K;       // one full A row per thread
    const __bf16* gB = Bt + (size_t)(n0 + rowB) * K + halfB;

    v8f zero = {0.f, 0.f, 0.f, 0.f, 0.f, 0.f, 0.f, 0.f};
    v8f acc[2][8];
#pragma unroll
    for (int tr = 0; tr < 2; ++tr)
#pragma unroll
        for (int i = 0; i < 8; ++i) acc[tr][i] = zero;

    // prologue: stage k-slab 0 into buffer 0
#pragma unroll
    for (int s = 0; s < 4; ++s) copy16(&As[0][t][s * 8], gA + s * 8);
    copy16(&Bs[0][rowB][halfB],     gB);
    copy16(&Bs[0][rowB][halfB + 8], gB + 8);
    copy_wait();
    __syncthreads();

    int buf = 0;
    for (int kk = 0; kk < K; kk += 32) {
        if (kk + 32 < K) {   // stage next slab into the other buffer (async)
            const __bf16* pa = gA + kk + 32;
            const __bf16* pb = gB + kk + 32;
#pragma unroll
            for (int s = 0; s < 4; ++s) copy16(&As[buf ^ 1][t][s * 8], pa + s * 8);
            copy16(&Bs[buf ^ 1][rowB][halfB],     pb);
            copy16(&Bs[buf ^ 1][rowB][halfB + 8], pb + 8);
            __builtin_prefetch(pa + 32, 0, 0);
            __builtin_prefetch(pb + 32, 0, 0);
        }

        // compute on current buffer: wave owns rows [32w, 32w+32)
        v16bf af0 = ld_frag2(&As[buf][(w * 2 + 0) * 16 + cl][g * 8],
                             &As[buf][(w * 2 + 0) * 16 + cl][16 + g * 8]);
        v16bf af1 = ld_frag2(&As[buf][(w * 2 + 1) * 16 + cl][g * 8],
                             &As[buf][(w * 2 + 1) * 16 + cl][16 + g * 8]);
#pragma unroll
        for (int c = 0; c < 8; ++c) {
            v16bf bfrag = ld_frag2(&Bs[buf][c * 16 + cl][g * 8], &Bs[buf][c * 16 + cl][16 + g * 8]);
            acc[0][c] = wmma_bf16(af0, bfrag, acc[0][c]);
            acc[1][c] = wmma_bf16(af1, bfrag, acc[1][c]);
        }

        copy_wait();
        __syncthreads();
        buf ^= 1;
    }

#pragma unroll
    for (int tr = 0; tr < 2; ++tr) {
#pragma unroll
        for (int c = 0; c < 8; ++c) {
            int col = n0 + c * 16 + cl;
#pragma unroll
            for (int v = 0; v < 8; ++v) {
                int m = m0 + (w * 2 + tr) * 16 + v + 8 * g;
                float val = acc[tr][c][v] + bias[col];
                if (EPI == 0) {
                    int which = col >> 10;
                    int cc = col & 1023;
                    int h = cc >> 6, d = cc & 63;
                    int bidx = m >> 11, tpos = m & 2047;
                    size_t idx = (((size_t)(bidx * NHEAD + h)) * TSEQ + tpos) * HEADD + d;
                    if (which == 0)      qv[idx] = f2bf(val * 0.125f);  // fold 1/sqrt(64)
                    else if (which == 1) kv[idx] = f2bf(val);
                    else                 vv[idx] = f2bf(val);
                } else if (EPI == 2) {
                    float gl = 0.5f * val * (1.0f + erff(val * 0.70710678f));
                    outb[(size_t)m * N + col] = f2bf(gl);
                } else {
                    outf[(size_t)m * N + col] = val + resid[(size_t)m * N + col];
                }
            }
        }
    }
}

// ---------------- flash attention (causal), 64 q-rows per WG ----------------

__global__ __launch_bounds__(256) void attn_kernel(const __bf16* __restrict__ qb,
                                                   const __bf16* __restrict__ kb,
                                                   const __bf16* __restrict__ vb,
                                                   __bf16* __restrict__ attno) {
    __shared__ __bf16 Qs[64][72];
    __shared__ __bf16 Ks[64][72];
    __shared__ __bf16 Vt[64][72];   // transposed: Vt[d][k]
    __shared__ __bf16 Ps[64][72];
    __shared__ float  Ss[64][72];
    __shared__ float  mS[64], lS[64], rsS[64];

    const int t    = threadIdx.x;
    const int w    = t >> 5;
    const int lane = t & 31;
    const int g    = lane >> 4;
    const int cl   = lane & 15;
    const int wr   = w >> 1;          // output tile row 0..3
    const int wc0  = (w & 1) * 2;     // output tile col pair {0,1} or {2,3}

    const int bh = blockIdx.y;
    const int b  = bh >> 4;
    const int h  = bh & 15;
    const int q0 = blockIdx.x * 64;
    const size_t base = (size_t)bh * TSEQ * HEADD;

    const int srow = t >> 2;
    const int sseg = (t & 3) * 16;

    // stage Q tile
    {
        const __bf16* gq = qb + base + (size_t)(q0 + srow) * HEADD + sseg;
        *(uint4*)&Qs[srow][sseg]     = ((const uint4*)gq)[0];
        *(uint4*)&Qs[srow][sseg + 8] = ((const uint4*)gq)[1];
    }
    if (t < 64) { mS[t] = -INFINITY; lS[t] = 0.f; }

    v8f zero = {0.f, 0.f, 0.f, 0.f, 0.f, 0.f, 0.f, 0.f};
    v8f o[2];
    o[0] = zero; o[1] = zero;
    __syncthreads();

    for (int kblk = 0; kblk <= blockIdx.x; ++kblk) {
        const int k0 = kblk * 64;
        // stage K (row-major) and V (transposed)
        {
            const __bf16* gk = kb + base + (size_t)(k0 + srow) * HEADD + sseg;
            *(uint4*)&Ks[srow][sseg]     = ((const uint4*)gk)[0];
            *(uint4*)&Ks[srow][sseg + 8] = ((const uint4*)gk)[1];
            const __bf16* gv = vb + base + (size_t)(k0 + srow) * HEADD + sseg;
            __bf16 tmp[16];
            *(uint4*)&tmp[0] = ((const uint4*)gv)[0];
            *(uint4*)&tmp[8] = ((const uint4*)gv)[1];
#pragma unroll
            for (int i = 0; i < 16; ++i) Vt[sseg + i][srow] = tmp[i];
        }
        __syncthreads();

        // S = Q * K^T  (two 16x16 tiles per wave, K-dim = 64 in two chunks)
        v8f s[2];
        s[0] = zero; s[1] = zero;
#pragma unroll
        for (int ch = 0; ch < 64; ch += 32) {
            v16bf afrag = ld_frag2(&Qs[wr * 16 + cl][ch + g * 8],
                                   &Qs[wr * 16 + cl][ch + 16 + g * 8]);
#pragma unroll
            for (int ci = 0; ci < 2; ++ci) {
                v16bf bfrag = ld_frag2(&Ks[(wc0 + ci) * 16 + cl][ch + g * 8], &Ks[(wc0 + ci) * 16 + cl][ch + 16 + g * 8]);
                s[ci] = wmma_bf16(afrag, bfrag, s[ci]);
            }
        }
        // causal mask + spill scores to LDS
#pragma unroll
        for (int ci = 0; ci < 2; ++ci) {
            int col = (wc0 + ci) * 16 + cl;
#pragma unroll
            for (int v = 0; v < 8; ++v) {
                int mloc = wr * 16 + v + 8 * g;
                float sv = s[ci][v];
                if ((k0 + col) > (q0 + mloc)) sv = -INFINITY;
                Ss[mloc][col] = sv;
            }
        }
        __syncthreads();

        // online softmax statistics: 4 threads per row, quad shfl reductions
        {
            int r  = t >> 2;
            int qp = (t & 3) * 16;
            float mold = mS[r];
            float mx = -INFINITY;
#pragma unroll
            for (int j = 0; j < 16; ++j) mx = fmaxf(mx, Ss[r][qp + j]);
            mx = fmaxf(mx, __shfl_xor(mx, 1, 32));
            mx = fmaxf(mx, __shfl_xor(mx, 2, 32));
            mx = fmaxf(mx, mold);
            float sum = 0.f;
#pragma unroll
            for (int j = 0; j < 16; ++j) {
                float p = __expf(Ss[r][qp + j] - mx);
                sum += p;
                Ps[r][qp + j] = f2bf(p);
            }
            sum += __shfl_xor(sum, 1, 32);
            sum += __shfl_xor(sum, 2, 32);
            if ((t & 3) == 0) {
                float rsc = __expf(mold - mx);
                lS[r] = lS[r] * rsc + sum;
                mS[r] = mx;
                rsS[r] = rsc;
            }
        }
        __syncthreads();

        // rescale O, then O += P * V
#pragma unroll
        for (int ti = 0; ti < 2; ++ti)
#pragma unroll
            for (int v = 0; v < 8; ++v) o[ti][v] *= rsS[wr * 16 + v + 8 * g];
#pragma unroll
        for (int ch = 0; ch < 64; ch += 32) {
            v16bf afrag = ld_frag2(&Ps[wr * 16 + cl][ch + g * 8],
                                   &Ps[wr * 16 + cl][ch + 16 + g * 8]);
#pragma unroll
            for (int ti = 0; ti < 2; ++ti) {
                v16bf bfrag = ld_frag2(&Vt[(wc0 + ti) * 16 + cl][ch + g * 8], &Vt[(wc0 + ti) * 16 + cl][ch + 16 + g * 8]);
                o[ti] = wmma_bf16(afrag, bfrag, o[ti]);
            }
        }
        __syncthreads();
    }

    // epilogue: divide by l, write bf16 to [B*T, EMB]
#pragma unroll
    for (int ti = 0; ti < 2; ++ti) {
        int dcol = (wc0 + ti) * 16 + cl;
#pragma unroll
        for (int v = 0; v < 8; ++v) {
            int mloc = wr * 16 + v + 8 * g;
            float val = o[ti][v] / lS[mloc];
            attno[(size_t)(b * TSEQ + q0 + mloc) * EMB + h * HEADD + dcol] = f2bf(val);
        }
    }
}

// ---------------- launch ----------------

extern "C" void kernel_launch(void* const* d_in, const int* in_sizes, int n_in,
                              void* d_out, int out_size, void* d_ws, size_t ws_size,
                              hipStream_t stream) {
    (void)in_sizes; (void)n_in; (void)out_size; (void)ws_size;
    const float* x      = (const float*)d_in[0];
    const float* ln1_w  = (const float*)d_in[1];
    const float* ln1_b  = (const float*)d_in[2];
    const float* w_attn = (const float*)d_in[3];
    const float* b_attn = (const float*)d_in[4];
    const float* w_proj = (const float*)d_in[5];
    const float* b_proj = (const float*)d_in[6];
    const float* ln2_w  = (const float*)d_in[7];
    const float* ln2_b  = (const float*)d_in[8];
    const float* w_fc1  = (const float*)d_in[9];
    const float* b_fc1  = (const float*)d_in[10];
    const float* w_fc2  = (const float*)d_in[11];
    const float* b_fc2  = (const float*)d_in[12];
    float* out = (float*)d_out;

    char* ws = (char*)d_ws;
    size_t off = 0;
    auto take = [&](size_t bytes) -> char* {
        char* p = ws + off;
        off = (off + bytes + 255) & ~(size_t)255;
        return p;
    };
    __bf16* wqkvT = (__bf16*)take((size_t)3 * EMB * EMB * 2);     // [3072][1024]
    __bf16* wprojT= (__bf16*)take((size_t)EMB * EMB * 2);         // [1024][1024]
    __bf16* wfc1T = (__bf16*)take((size_t)FFD * EMB * 2);         // [4096][1024]
    __bf16* wfc2T = (__bf16*)take((size_t)EMB * FFD * 2);         // [1024][4096]
    __bf16* aln   = (__bf16*)take((size_t)BT * EMB * 2);          // LN1 out, reused for LN2 out
    __bf16* qv    = (__bf16*)take((size_t)BT * EMB * 2);          // [B,H,T,D]
    __bf16* kv    = (__bf16*)take((size_t)BT * EMB * 2);
    __bf16* vv    = (__bf16*)take((size_t)BT * EMB * 2);
    __bf16* attno = (__bf16*)take((size_t)BT * EMB * 2);          // [B*T, EMB]
    float*  x1    = (float*) take((size_t)BT * EMB * 4);          // residual stream after attn
    __bf16* ff1   = (__bf16*)take((size_t)BT * FFD * 2);          // GELU output

    // weight conversion + transpose
    {
        size_t n;
        n = (size_t)EMB * 3 * EMB;
        wconv_t<<<dim3((unsigned)((n + 255) / 256)), 256, 0, stream>>>(w_attn, wqkvT, EMB, 3 * EMB);
        n = (size_t)EMB * EMB;
        wconv_t<<<dim3((unsigned)((n + 255) / 256)), 256, 0, stream>>>(w_proj, wprojT, EMB, EMB);
        n = (size_t)EMB * FFD;
        wconv_t<<<dim3((unsigned)((n + 255) / 256)), 256, 0, stream>>>(w_fc1, wfc1T, EMB, FFD);
        n = (size_t)FFD * EMB;
        wconv_t<<<dim3((unsigned)((n + 255) / 256)), 256, 0, stream>>>(w_fc2, wfc2T, FFD, EMB);
    }

    // LN1
    ln_kernel<<<dim3(BT), 256, 0, stream>>>(x, ln1_w, ln1_b, aln);

    // QKV projection (M=8192, N=3072, K=1024), scatter into [B,H,T,D]
    gemm_bf16<0><<<dim3(3 * EMB / 128, BT / 256), 256, 0, stream>>>(
        aln, wqkvT, b_attn, nullptr, nullptr, nullptr, qv, kv, vv, BT, 3 * EMB, EMB);

    // causal flash attention
    attn_kernel<<<dim3(TSEQ / 64, BBATCH * NHEAD), 256, 0, stream>>>(qv, kv, vv, attno);

    // output projection + residual (M=8192, N=1024, K=1024) -> x1 (f32)
    gemm_bf16<1><<<dim3(EMB / 128, BT / 256), 256, 0, stream>>>(
        attno, wprojT, b_proj, x, x1, nullptr, nullptr, nullptr, nullptr, BT, EMB, EMB);

    // LN2
    ln_kernel<<<dim3(BT), 256, 0, stream>>>(x1, ln2_w, ln2_b, aln);

    // FC1 + GELU (M=8192, N=4096, K=1024) -> ff1 (bf16)
    gemm_bf16<2><<<dim3(FFD / 128, BT / 256), 256, 0, stream>>>(
        aln, wfc1T, b_fc1, nullptr, nullptr, ff1, nullptr, nullptr, nullptr, BT, FFD, EMB);

    // FC2 + residual (M=8192, N=1024, K=4096) -> out (f32)
    gemm_bf16<3><<<dim3(EMB / 128, BT / 256), 256, 0, stream>>>(
        ff1, wfc2T, b_fc2, x1, out, nullptr, nullptr, nullptr, nullptr, BT, EMB, FFD);
}
